// RephineLayer_Equiv_18107582120178
// MI455X (gfx1250) — compile-verified, matrix operand order checked by reading, not run on hardware
//
#include <hip/hip_runtime.h>
#include <hip/hip_bf16.h>
#include <math.h>

// ---------------------------------------------------------------------------
// RephineLayer surrogate for MI455X (gfx1250, wave32, WMMA).
// Heavy GEMMs (vertex/edge MLP layer 1) run on v_wmma_f32_16x16x32_f16 with
// per-fragment pre-swizzled weights in LDS (contiguous 32B per lane ->
// ds_load_b128 pairs). B-fragments for a K-chunk are batch-loaded before the
// WMMA quartet so the DScnt wait drains once. Scatter min/max over positive
// sigmoid outputs uses u32 atomics on float bits. dim-1 term collapses to a
// constant vector -> computed on chip.
// ---------------------------------------------------------------------------

typedef _Float16 half_t;
typedef __attribute__((ext_vector_type(16))) _Float16 v16h;
typedef __attribute__((ext_vector_type(8)))  float    v8f;

#define WAVES_PER_BLOCK 8
#define BLOCK_THREADS   (WAVES_PER_BLOCK * 32)
#define ROWS_PER_WAVE   16
#define ROWS_PER_BLOCK  (WAVES_PER_BLOCK * ROWS_PER_WAVE)

__device__ __forceinline__ float fast_sigmoid(float x) {
    return 1.0f / (1.0f + __expf(-x));
}

// Fill the per-fragment swizzled B matrix (K=128 x N=64, f16) in LDS.
// Fragment (kc, nt): lane l holds, for column n=(l&15)+nt*16,
//   K = kc*32 + (l>>4)*8 + j        for j in 0..7
//   K = kc*32 + (l>>4)*8 + 8 + j    for j in 8..15   (i.e. +16 over group 0)
__device__ __forceinline__ void fill_bswz(half_t* bswz, const float* W, int tid) {
    for (int idx = tid; idx < 4 * 4 * 32 * 16; idx += BLOCK_THREADS) {
        const int j    = idx & 15;
        const int lane = (idx >> 4) & 31;
        const int nt   = (idx >> 9) & 3;
        const int kc   = (idx >> 11) & 3;
        const int n    = (lane & 15) + nt * 16;
        const int K    = kc * 32 + (lane >> 4) * 8 + ((j < 8) ? j : (8 + j));
        bswz[idx] = (half_t)W[K * 64 + n];
    }
}

__device__ __forceinline__ v16h cvt_a_frag(const float4* r0, const float4* r1) {
    // r0: 8 floats at K-group 0, r1: 8 floats at K-group 1 (K+16)
    v16h a;
    float4 p0 = r0[0], p1 = r0[1], q0 = r1[0], q1 = r1[1];
    a[0] = (half_t)p0.x; a[1] = (half_t)p0.y; a[2] = (half_t)p0.z; a[3] = (half_t)p0.w;
    a[4] = (half_t)p1.x; a[5] = (half_t)p1.y; a[6] = (half_t)p1.z; a[7] = (half_t)p1.w;
    a[8]  = (half_t)q0.x; a[9]  = (half_t)q0.y; a[10] = (half_t)q0.z; a[11] = (half_t)q0.w;
    a[12] = (half_t)q1.x; a[13] = (half_t)q1.y; a[14] = (half_t)q1.z; a[15] = (half_t)q1.w;
    return a;
}

// ---------------------------------------------------------------------------
// Kernel 0: workspace init (dmin bits = +inf, dmax bits = 0, BN sums = 0)
// ---------------------------------------------------------------------------
__global__ void init_ws_kernel(unsigned* __restrict__ dminb,
                               unsigned* __restrict__ dmaxb,
                               float* __restrict__ bnsum,
                               float* __restrict__ bnsum2,
                               long long nF) {
    long long i = (long long)blockIdx.x * blockDim.x + threadIdx.x;
    long long stride = (long long)gridDim.x * blockDim.x;
    for (long long k = i; k < nF; k += stride) {
        dminb[k] = 0x7F800000u;  // +inf
        dmaxb[k] = 0u;           // < any positive float
    }
    if (i < 64) { bnsum[i] = 0.0f; bnsum2[i] = 0.0f; }
}

// ---------------------------------------------------------------------------
// Kernel 1: vertex filtration  fv = sigmoid(relu(x@W1+b1)@W2+b2)   [N,8]
// ---------------------------------------------------------------------------
__global__ __launch_bounds__(BLOCK_THREADS)
void vertex_mlp_kernel(const float* __restrict__ x,
                       const float* __restrict__ W1, const float* __restrict__ b1,
                       const float* __restrict__ W2, const float* __restrict__ b2,
                       float* __restrict__ fv) {
    __shared__ __align__(32) half_t bswz[4 * 4 * 32 * 16];       // 16 KB
    __shared__ float  lw2t[8][64];                               // W2 transposed [f][k]
    __shared__ float  lb1[64];
    __shared__ float  lb2[8];
    __shared__ __align__(16) float hbuf[WAVES_PER_BLOCK][16][68]; // relu(h1)

    const int t = threadIdx.x;
    fill_bswz(bswz, W1, t);
    for (int i = t; i < 64 * 8; i += BLOCK_THREADS) lw2t[i & 7][i >> 3] = W2[i];
    if (t < 64) lb1[t] = b1[t];
    if (t < 8)  lb2[t] = b2[t];
    __syncthreads();

    const int wave = t >> 5;
    const int lane = t & 31;
    const int m    = lane & 15;     // row within 16-row tile
    const int ksel = lane >> 4;     // half-wave selects K sub-group
    const int row  = (blockIdx.x * WAVES_PER_BLOCK + wave) * ROWS_PER_WAVE + m;
    const float* xr = x + (size_t)row * 128;

    // Prefetch the next tile's rows (global_prefetch_b8 path).
    __builtin_prefetch(xr + (size_t)ROWS_PER_BLOCK * 128, 0, 3);

    v8f acc[4];
    #pragma unroll
    for (int nt = 0; nt < 4; ++nt)
        #pragma unroll
        for (int v = 0; v < 8; ++v) acc[nt][v] = 0.0f;

    #pragma unroll
    for (int kc = 0; kc < 4; ++kc) {          // K = 128 in chunks of 32
        const int k0 = kc * 32 + ksel * 8;
        const v16h a = cvt_a_frag((const float4*)(xr + k0),
                                  (const float4*)(xr + k0 + 16));
        // Batch-load all 4 B fragments for this K-chunk, then issue WMMAs.
        v16h bfr[4];
        #pragma unroll
        for (int nt = 0; nt < 4; ++nt)
            bfr[nt] = *(const v16h*)&bswz[((kc * 4 + nt) * 32 + lane) * 16];
        #pragma unroll
        for (int nt = 0; nt < 4; ++nt)
            acc[nt] = __builtin_amdgcn_wmma_f32_16x16x32_f16(
                false, a, false, bfr[nt], (short)0, acc[nt], false, false);
    }

    // bias + relu -> LDS tile.  C/D layout: element v of lane -> (M=v+8*ksel, N=lane&15)
    #pragma unroll
    for (int nt = 0; nt < 4; ++nt) {
        const int n = (lane & 15) + nt * 16;
        #pragma unroll
        for (int v = 0; v < 8; ++v) {
            const int mm = v + 8 * ksel;
            hbuf[wave][mm][n] = fmaxf(acc[nt][v] + lb1[n], 0.0f);
        }
    }
    __syncthreads();

    // layer 2: lane -> row m, filtrations [ksel*4, ksel*4+4). All b128 LDS reads.
    float4 hrow[16];
    const float4* hb4 = (const float4*)&hbuf[wave][m][0];
    #pragma unroll
    for (int kk = 0; kk < 16; ++kk) hrow[kk] = hb4[kk];

    const int f0 = ksel * 4;
    float4 res;
    float* resp = (float*)&res;
    #pragma unroll
    for (int fi = 0; fi < 4; ++fi) {
        const int f = f0 + fi;
        float s = lb2[f];
        const float4* w4 = (const float4*)&lw2t[f][0];
        #pragma unroll
        for (int kk = 0; kk < 16; ++kk) {
            const float4 wv = w4[kk];
            s += hrow[kk].x * wv.x + hrow[kk].y * wv.y +
                 hrow[kk].z * wv.z + hrow[kk].w * wv.w;
        }
        resp[fi] = fast_sigmoid(s);
    }
    *(float4*)&fv[(size_t)row * 8 + f0] = res;   // 16B-aligned b128 store
}

// ---------------------------------------------------------------------------
// Kernel 2: edge filtration + scatter min/max.
//   fe = sigmoid(relu((x[s]+x[d])@We1[0:128] + dist*We1[128] + be1)@We2+be2)
// ---------------------------------------------------------------------------
__global__ __launch_bounds__(BLOCK_THREADS)
void edge_mlp_kernel(const float* __restrict__ x,
                     const float* __restrict__ pos,
                     const int* __restrict__ srcI, const int* __restrict__ dstI,
                     const float* __restrict__ We1, const float* __restrict__ be1,
                     const float* __restrict__ We2, const float* __restrict__ be2,
                     unsigned* __restrict__ dminb, unsigned* __restrict__ dmaxb) {
    __shared__ __align__(32) half_t bswz[4 * 4 * 32 * 16];       // 16 KB
    __shared__ float  lwrow[64];                // We1 row 128 (dist column weights)
    __shared__ float  lw2t[8][64];
    __shared__ float  lb1[64];
    __shared__ float  lb2[8];
    __shared__ __align__(16) float hbuf[WAVES_PER_BLOCK][16][68];

    const int t = threadIdx.x;
    fill_bswz(bswz, We1, t);                    // rows 0..127 only
    if (t < 64) { lwrow[t] = We1[128 * 64 + t]; lb1[t] = be1[t]; }
    for (int i = t; i < 64 * 8; i += BLOCK_THREADS) lw2t[i & 7][i >> 3] = We2[i];
    if (t < 8) lb2[t] = be2[t];
    __syncthreads();

    const int wave = t >> 5;
    const int lane = t & 31;
    const int m    = lane & 15;
    const int ksel = lane >> 4;
    const int e    = (blockIdx.x * WAVES_PER_BLOCK + wave) * ROWS_PER_WAVE + m;

    const int s = srcI[e];
    const int d = dstI[e];
    const float* xs = x + (size_t)s * 128;
    const float* xd = x + (size_t)d * 128;

    // Every lane computes the dist for its row (duplicated across half-waves);
    // row values are fetched later with intra-wave shuffles (no LDS round-trip).
    const float dx = pos[(size_t)s * 3 + 0] - pos[(size_t)d * 3 + 0];
    const float dy = pos[(size_t)s * 3 + 1] - pos[(size_t)d * 3 + 1];
    const float dz = pos[(size_t)s * 3 + 2] - pos[(size_t)d * 3 + 2];
    const float mydist = sqrtf(dx * dx + dy * dy + dz * dz);

    v8f acc[4];
    #pragma unroll
    for (int nt = 0; nt < 4; ++nt)
        #pragma unroll
        for (int v = 0; v < 8; ++v) acc[nt][v] = 0.0f;

    #pragma unroll
    for (int kc = 0; kc < 4; ++kc) {
        const int k0 = kc * 32 + ksel * 8;
        // gather + sum both endpoints (b128 loads), convert to f16
        const float4* s0 = (const float4*)(xs + k0);
        const float4* s1 = (const float4*)(xs + k0 + 16);
        const float4* d0 = (const float4*)(xd + k0);
        const float4* d1 = (const float4*)(xd + k0 + 16);
        float4 sum0 = s0[0], sum0b = s0[1], sum1 = s1[0], sum1b = s1[1];
        float4 t0 = d0[0], t0b = d0[1], t1 = d1[0], t1b = d1[1];
        sum0.x += t0.x;  sum0.y += t0.y;  sum0.z += t0.z;  sum0.w += t0.w;
        sum0b.x += t0b.x; sum0b.y += t0b.y; sum0b.z += t0b.z; sum0b.w += t0b.w;
        sum1.x += t1.x;  sum1.y += t1.y;  sum1.z += t1.z;  sum1.w += t1.w;
        sum1b.x += t1b.x; sum1b.y += t1b.y; sum1b.z += t1b.z; sum1b.w += t1b.w;
        v16h a;
        a[0] = (half_t)sum0.x;  a[1] = (half_t)sum0.y;  a[2] = (half_t)sum0.z;  a[3] = (half_t)sum0.w;
        a[4] = (half_t)sum0b.x; a[5] = (half_t)sum0b.y; a[6] = (half_t)sum0b.z; a[7] = (half_t)sum0b.w;
        a[8]  = (half_t)sum1.x;  a[9]  = (half_t)sum1.y;  a[10] = (half_t)sum1.z;  a[11] = (half_t)sum1.w;
        a[12] = (half_t)sum1b.x; a[13] = (half_t)sum1b.y; a[14] = (half_t)sum1b.z; a[15] = (half_t)sum1b.w;

        // Batch-load all 4 B fragments for this K-chunk, then issue WMMAs.
        v16h bfr[4];
        #pragma unroll
        for (int nt = 0; nt < 4; ++nt)
            bfr[nt] = *(const v16h*)&bswz[((kc * 4 + nt) * 32 + lane) * 16];
        #pragma unroll
        for (int nt = 0; nt < 4; ++nt)
            acc[nt] = __builtin_amdgcn_wmma_f32_16x16x32_f16(
                false, a, false, bfr[nt], (short)0, acc[nt], false, false);
    }

    // dist values for rows mm = v + 8*ksel via wave shuffle (lane mm holds row mm).
    float distv[8];
    #pragma unroll
    for (int v = 0; v < 8; ++v) distv[v] = __shfl(mydist, v + 8 * ksel, 32);

    #pragma unroll
    for (int nt = 0; nt < 4; ++nt) {
        const int n = (lane & 15) + nt * 16;
        #pragma unroll
        for (int v = 0; v < 8; ++v) {
            const int mm = v + 8 * ksel;
            const float h = acc[nt][v] + distv[v] * lwrow[n] + lb1[n];
            hbuf[wave][mm][n] = fmaxf(h, 0.0f);
        }
    }
    __syncthreads();

    // layer 2 + scatter: lane -> (row m, filtrations [ksel*4, ksel*4+4))
    float4 hrow[16];
    const float4* hb4 = (const float4*)&hbuf[wave][m][0];
    #pragma unroll
    for (int kk = 0; kk < 16; ++kk) hrow[kk] = hb4[kk];

    const int f0 = ksel * 4;
    #pragma unroll
    for (int fi = 0; fi < 4; ++fi) {
        const int f = f0 + fi;
        float sum = lb2[f];
        const float4* w4 = (const float4*)&lw2t[f][0];
        #pragma unroll
        for (int kk = 0; kk < 16; ++kk) {
            const float4 wv = w4[kk];
            sum += hrow[kk].x * wv.x + hrow[kk].y * wv.y +
                   hrow[kk].z * wv.z + hrow[kk].w * wv.w;
        }
        const unsigned bits = __float_as_uint(fast_sigmoid(sum));  // > 0 always
        atomicMin(&dminb[(size_t)s * 8 + f], bits);
        atomicMin(&dminb[(size_t)d * 8 + f], bits);
        atomicMax(&dmaxb[(size_t)s * 8 + f], bits);
        atomicMax(&dmaxb[(size_t)d * 8 + f], bits);
    }
}

// ---------------------------------------------------------------------------
// Kernel 3: per-graph  lin0 = pers0@Wd0 (K effective 24), mean over 400 verts,
//           + constant x1 (= Wd1 rows 1..3 sum + bd1), BN stat accumulation.
// ---------------------------------------------------------------------------
__global__ __launch_bounds__(256)
void graph_reduce_kernel(const float* __restrict__ fv,
                         const unsigned* __restrict__ dminb,
                         const unsigned* __restrict__ dmaxb,
                         const float* __restrict__ Wd0, const float* __restrict__ bd0,
                         const float* __restrict__ Wd1, const float* __restrict__ bd1,
                         float* __restrict__ hout,
                         float* __restrict__ bnsum, float* __restrict__ bnsum2,
                         int vPer, int nGraphs) {
    __shared__ float lwd[24][64];     // Wd0 rows remapped: c -> (f=c%8, slot 1/2/3)
    __shared__ float in24[64][25];    // 64 vertices x 24 inputs (padded)
    __shared__ float hsum[4][64];

    const int g = blockIdx.x;
    const int t = threadIdx.x;

    for (int i = t; i < 24 * 64; i += 256) {
        const int c = i >> 6, j = i & 63;
        int f, slot;
        if (c < 8)       { f = c;      slot = 1; }   // fv
        else if (c < 16) { f = c - 8;  slot = 2; }   // gamma (dmax)
        else             { f = c - 16; slot = 3; }   // death (dmin)
        lwd[c][j] = Wd0[(4 * f + slot) * 64 + j];
    }
    hsum[t >> 6][t & 63] = 0.0f;
    __syncthreads();

    const int j  = t & 63;
    const int vg = t >> 6;
    float acc = 0.0f;
    const long long base = (long long)g * vPer;

    for (int c0 = 0; c0 < vPer; c0 += 64) {
        const int cn = min(64, vPer - c0);
        __syncthreads();
        for (int i = t; i < cn * 24; i += 256) {
            const int vi = i / 24, c = i % 24;
            const long long v = base + c0 + vi;
            float val;
            if (c < 8) {
                val = fv[v * 8 + c];
            } else if (c < 16) {
                const unsigned b = dmaxb[v * 8 + (c - 8)];
                val = (b == 0u) ? 1.0f : __uint_as_float(b);          // isolated -> 1
            } else {
                const unsigned b = dminb[v * 8 + (c - 16)];
                val = (b == 0x7F800000u) ? 1.0f : __uint_as_float(b); // isolated -> 1
            }
            in24[vi][c] = val;
        }
        __syncthreads();
        for (int vi = vg; vi < cn; vi += 4) {
            float srow = 0.0f;
            #pragma unroll
            for (int c = 0; c < 24; ++c) srow += in24[vi][c] * lwd[c][j];
            acc += srow;
        }
    }
    hsum[vg][j] = acc;
    __syncthreads();

    if (t < 64) {
        const float tot = hsum[0][t] + hsum[1][t] + hsum[2][t] + hsum[3][t];
        const float x0 = tot / (float)vPer + bd0[t];
        const float x1 = Wd1[64 + t] + Wd1[128 + t] + Wd1[192 + t] + bd1[t];
        const float h  = x0 + x1;
        hout[(size_t)g * 64 + t] = h;
        atomicAdd(&bnsum[t],  h);
        atomicAdd(&bnsum2[t], h * h);
        (void)nGraphs;
    }
}

// ---------------------------------------------------------------------------
// Kernel 4: batch-norm (batch stats over G) + out MLP -> d_out [G,64]
// ---------------------------------------------------------------------------
__global__ __launch_bounds__(64)
void bn_out_kernel(const float* __restrict__ hbuf,
                   const float* __restrict__ bnsum, const float* __restrict__ bnsum2,
                   const float* __restrict__ gamma, const float* __restrict__ beta,
                   const float* __restrict__ Wo1, const float* __restrict__ bo1,
                   const float* __restrict__ Wo2, const float* __restrict__ bo2,
                   float* __restrict__ out, int nGraphs) {
    __shared__ float hn[64];
    __shared__ float t1[64];
    const int g = blockIdx.x;
    const int t = threadIdx.x;

    const float invG = 1.0f / (float)nGraphs;
    const float mu  = bnsum[t] * invG;
    const float var = bnsum2[t] * invG - mu * mu;
    const float h   = hbuf[(size_t)g * 64 + t];
    hn[t] = (h - mu) * rsqrtf(var + 1e-5f) * gamma[t] + beta[t];
    __syncthreads();

    float s = bo1[t];
    #pragma unroll
    for (int k = 0; k < 64; ++k) s += hn[k] * Wo1[k * 64 + t];
    t1[t] = fmaxf(s, 0.0f);
    __syncthreads();

    float o = bo2[t];
    #pragma unroll
    for (int k = 0; k < 64; ++k) o += t1[k] * Wo2[k * 64 + t];
    out[(size_t)g * 64 + t] = o;
}

// ---------------------------------------------------------------------------
// Launch
// ---------------------------------------------------------------------------
extern "C" void kernel_launch(void* const* d_in, const int* in_sizes, int n_in,
                              void* d_out, int out_size, void* d_ws, size_t ws_size,
                              hipStream_t stream) {
    const float* x    = (const float*)d_in[0];
    const float* pos  = (const float*)d_in[1];
    const int*   ei   = (const int*)d_in[2];
    const float* W1   = (const float*)d_in[6];
    const float* b1   = (const float*)d_in[7];
    const float* W2   = (const float*)d_in[8];
    const float* b2   = (const float*)d_in[9];
    const float* We1  = (const float*)d_in[10];
    const float* be1  = (const float*)d_in[11];
    const float* We2  = (const float*)d_in[12];
    const float* be2  = (const float*)d_in[13];
    const float* Wd0  = (const float*)d_in[14];
    const float* bd0  = (const float*)d_in[15];
    const float* Wd1  = (const float*)d_in[16];
    const float* bd1  = (const float*)d_in[17];
    const float* bng  = (const float*)d_in[18];
    const float* bnb  = (const float*)d_in[19];
    const float* Wo1  = (const float*)d_in[20];
    const float* bo1  = (const float*)d_in[21];
    const float* Wo2  = (const float*)d_in[22];
    const float* bo2  = (const float*)d_in[23];
    float* out = (float*)d_out;

    const long long N = in_sizes[0] / 128;      // 204800
    const long long E = in_sizes[2] / 2;        // 409600
    const int G = in_sizes[3] - 1;              // 512
    const int vPer = (int)(N / G);              // 400
    const int* srcI = ei;
    const int* dstI = ei + E;

    // Workspace layout
    float*    fv     = (float*)d_ws;                       // N*8 f32
    unsigned* dminb  = (unsigned*)(fv + N * 8);            // N*8 u32
    unsigned* dmaxb  = dminb + N * 8;                      // N*8 u32
    float*    hbuf   = (float*)(dmaxb + N * 8);            // G*64 f32
    float*    bnsum  = hbuf + (size_t)G * 64;              // 64
    float*    bnsum2 = bnsum + 64;                         // 64
    (void)ws_size; (void)n_in; (void)out_size;

    const long long nF = N * 8;
    init_ws_kernel<<<1024, 256, 0, stream>>>(dminb, dmaxb, bnsum, bnsum2, nF);

    const int vBlocks = (int)(N / ROWS_PER_BLOCK);         // 1600
    vertex_mlp_kernel<<<vBlocks, BLOCK_THREADS, 0, stream>>>(x, W1, b1, W2, b2, fv);

    const int eBlocks = (int)(E / ROWS_PER_BLOCK);         // 3200
    edge_mlp_kernel<<<eBlocks, BLOCK_THREADS, 0, stream>>>(
        x, pos, srcI, dstI, We1, be1, We2, be2, dminb, dmaxb);

    graph_reduce_kernel<<<G, 256, 0, stream>>>(
        fv, dminb, dmaxb, Wd0, bd0, Wd1, bd1, hbuf, bnsum, bnsum2, vPer, G);

    bn_out_kernel<<<G, 64, 0, stream>>>(
        hbuf, bnsum, bnsum2, bng, bnb, Wo1, bo1, Wo2, bo2, out, G);
}
